// LSTM_68985764708483
// MI455X (gfx1250) — matrix operand order repaired and checked by reading, not compiled, not run on hardware
//
#include <hip/hip_runtime.h>
#include <hip/hip_fp16.h>

typedef __attribute__((ext_vector_type(16))) _Float16 v16h;
typedef __attribute__((ext_vector_type(8)))  float    v8f;

union U16h { v16h v; uint4 q[2]; };

__device__ __forceinline__ float fsig(float x) {
    return 1.0f / (1.0f + __expf(-x));
}
__device__ __forceinline__ float ftanh(float x) {
    x = fminf(fmaxf(x, -12.0f), 12.0f);
    float e = __expf(2.0f * x);
    return (e - 1.0f) / (e + 1.0f);
}

// ---------------------------------------------------------------------------
// Weight prep: fp16 combined weight Wc (K x 1024) in WMMA B-operand tiles.
//   K = Kin + 256.  Rows [0,Kin) = Wih^T (zero-padded past Iw), rows [Kin,K) = Whh^T.
//   Column permutation: ncolp = hb*64 + gate*16 + n  <->  4H row = gate*256 + hb*16 + n.
//   Tile (kt, nt): 32(K) x 16(N), stored at (nt*KT + kt)*512 elems; lane l owns the 16
//   contiguous fp16 [l*16, l*16+16):  elem kk = Wc[kt*32 + (l>>4)*8 + (kk&7) + (kk>>3)*16][nt*16 + (l&15)].
// Also combined bias bc[1024] in permuted column order.
// ---------------------------------------------------------------------------
__global__ void prep_wc_kernel(const float* __restrict__ Wih, const float* __restrict__ Whh,
                               const float* __restrict__ bih, const float* __restrict__ bhh,
                               _Float16* __restrict__ Wc, float* __restrict__ bc,
                               int Kin, int KT, int Iw, int total) {
    int idx = blockIdx.x * blockDim.x + threadIdx.x;
    if (idx < total) {
        int tile   = idx >> 9;
        int within = idx & 511;
        int lane   = within >> 4;
        int kk     = within & 15;
        int nt = tile / KT;
        int kt = tile % KT;
        int k = kt * 32 + ((lane >> 4) * 8) + (kk & 7) + ((kk >> 3) * 16);
        int ncolp = nt * 16 + (lane & 15);
        int hb = ncolp >> 6;
        int g  = (ncolp >> 4) & 3;
        int n  = ncolp & 15;
        int row = g * 256 + hb * 16 + n;           // row in 4H dim
        float wv;
        if (k < Kin) {
            wv = (k < Iw) ? Wih[row * Iw + k] : 0.0f;
        } else {
            wv = Whh[row * 256 + (k - Kin)];
        }
        Wc[idx] = (_Float16)wv;
    }
    if (idx < 1024) {
        int hb = idx >> 6;
        int g  = (idx >> 4) & 3;
        int n  = idx & 15;
        int row = g * 256 + hb * 16 + n;
        bc[idx] = bih[row] + bhh[row];
    }
}

// W_out (8 x 256) -> padded fp16 B-operand tiles Wp (256K x 16N), 8 k-tiles.
__global__ void prep_wout_kernel(const float* __restrict__ Wout, _Float16* __restrict__ Wp) {
    int idx = blockIdx.x * blockDim.x + threadIdx.x;   // 8*512 = 4096
    if (idx >= 4096) return;
    int kt = idx >> 9;
    int within = idx & 511;
    int lane = within >> 4;
    int kk = within & 15;
    int k = kt * 32 + ((lane >> 4) * 8) + (kk & 7) + ((kk >> 3) * 16);
    int n = lane & 15;
    Wp[idx] = (_Float16)((n < 8) ? Wout[n * 256 + k] : 0.0f);
}

// ---------------------------------------------------------------------------
// Fused LSTM layer. One WG = 32 batch rows, 8 wave32s, loops t=0..95.
//   gates = [x_t | h_{t-1}] @ Wc + bc, gate-interleaved columns.
//   Wave w owns 4 (m,hb) pairs; c stays in VGPRs across all 96 steps.
//   h double-buffered in LDS (parity of t): compute reads hbuf[pb], writes hbuf[pb^1].
//   LAYER==0: h streamed to global fp16 buffer.
//   LAYER==1: for t>=63 waves 0..1 do a WMMA projection (h @ WoutT + b_out) -> d_out.
// ---------------------------------------------------------------------------
template<int KIN, int KT, int LAYER>
__global__ __launch_bounds__(256)
void lstm_layer_kernel(const float* __restrict__ xinit,   // layer0: (B,64,8)
                       const float* __restrict__ xnext,   // layer0: (B,32,8)
                       const _Float16* __restrict__ h0in, // layer1: (B,96,256) fp16
                       const _Float16* __restrict__ Wc,
                       const float* __restrict__ bc,
                       _Float16* __restrict__ h0out,      // layer0 output
                       const _Float16* __restrict__ Wp,   // layer1: tiled W_out
                       const float* __restrict__ bout,    // layer1: (8)
                       float* __restrict__ out)           // (B,33,8)
{
    constexpr int KTX = KIN / 32;            // k-tiles covering x part
    __shared__ _Float16 xbuf[32 * KIN];
    __shared__ _Float16 hbuf[2][32 * 256];

    const int tid = threadIdx.x;
    const int w = tid >> 5;
    const int l = tid & 31;
    const int bbase = blockIdx.x * 32;
    const int lr = l & 15;                   // lane row/col within tile
    const int k0b = (l >> 4) * 8;            // A-fragment K sub-offset

    for (int i = tid; i < 32 * KIN; i += 256) xbuf[i] = (_Float16)0.0f;
    for (int i = tid; i < 2 * 32 * 256; i += 256) (&hbuf[0][0])[i] = (_Float16)0.0f;

    v8f c[4];
#pragma unroll
    for (int p = 0; p < 4; ++p)
#pragma unroll
        for (int e = 0; e < 8; ++e) c[p][e] = 0.0f;

    __syncthreads();

    for (int t = 0; t < 96; ++t) {
        const int pb = t & 1;
        const _Float16* hread = &hbuf[pb][0];
        _Float16* hwrite = &hbuf[pb ^ 1][0];

        // ---- stage x_t ----
        if (LAYER == 0) {
            int row = tid >> 3, ii = tid & 7;      // 32 rows x 8 cols (cols 8..31 stay 0)
            float xv = (t < 64)
                ? xinit[((size_t)(bbase + row) * 64 + t) * 8 + ii]
                : xnext[((size_t)(bbase + row) * 32 + (t - 64)) * 8 + ii];
            xbuf[row * KIN + ii] = (_Float16)xv;
        } else {
            int row = tid >> 3, off = (tid & 7) * 32;
            const uint4* src = (const uint4*)(h0in + ((size_t)(bbase + row) * 96 + t) * 256 + off);
            uint4* dst = (uint4*)(&xbuf[row * KIN + off]);
#pragma unroll
            for (int jj = 0; jj < 4; ++jj) dst[jj] = src[jj];
            if (t < 95)    // global_prefetch_b8 next step's slice
                __builtin_prefetch(h0in + ((size_t)(bbase + row) * 96 + (t + 1)) * 256 + off, 0, 0);
        }
        __syncthreads();
        // keep weight-fragment loads inside the t loop (no LICM -> scratch spills)
        asm volatile("" ::: "memory");

        // ---- WMMA gate GEMM + LSTM cell ----
#pragma unroll
        for (int p = 0; p < 4; ++p) {
            const int pair = w * 4 + p;
            const int m  = pair >> 4;            // 0..1
            const int hb = pair & 15;            // 0..15

            v8f acc[4];
#pragma unroll
            for (int g = 0; g < 4; ++g) {
                float bv = bc[hb * 64 + g * 16 + lr];
#pragma unroll
                for (int e = 0; e < 8; ++e) acc[g][e] = bv;
            }

            const int arow = m * 16 + lr;
            const _Float16* xrow = &xbuf[arow * KIN + k0b];
            const _Float16* hrow = hread + arow * 256 + k0b;

            // x part of K
            for (int kt = 0; kt < KTX; ++kt) {
                U16h ua;
                ua.q[0] = *(const uint4*)(xrow + kt * 32);
                ua.q[1] = *(const uint4*)(xrow + kt * 32 + 16);
#pragma unroll
                for (int g = 0; g < 4; ++g) {
                    const int nt = hb * 4 + g;
                    const _Float16* bp = Wc + (((size_t)(nt * KT + kt)) << 9) + l * 16;
                    U16h ub;
                    ub.q[0] = *(const uint4*)(bp);
                    ub.q[1] = *(const uint4*)(bp + 8);
                    acc[g] = __builtin_amdgcn_wmma_f32_16x16x32_f16(
                        false, ua.v, false, ub.v, (short)0, acc[g], false, false);
                }
            }
            // h part of K (8 tiles)
            for (int kt = 0; kt < 8; ++kt) {
                U16h ua;
                ua.q[0] = *(const uint4*)(hrow + kt * 32);
                ua.q[1] = *(const uint4*)(hrow + kt * 32 + 16);
#pragma unroll
                for (int g = 0; g < 4; ++g) {
                    const int nt = hb * 4 + g;
                    const _Float16* bp = Wc + (((size_t)(nt * KT + KTX + kt)) << 9) + l * 16;
                    U16h ub;
                    ub.q[0] = *(const uint4*)(bp);
                    ub.q[1] = *(const uint4*)(bp + 8);
                    acc[g] = __builtin_amdgcn_wmma_f32_16x16x32_f16(
                        false, ua.v, false, ub.v, (short)0, acc[g], false, false);
                }
            }

            // elementwise cell; write h_t straight into the other LDS buffer
            v8f cc = c[p];
            const int rb = m * 16 + ((l >> 4) * 8);
            const int col = hb * 16 + lr;
#pragma unroll
            for (int e = 0; e < 8; ++e) {
                float ig = fsig(acc[0][e]);
                float fg = fsig(acc[1][e]);
                float gg = ftanh(acc[2][e]);
                float og = fsig(acc[3][e]);
                float cn = fg * cc[e] + ig * gg;
                cc[e] = cn;
                hwrite[(rb + e) * 256 + col] = (_Float16)(og * ftanh(cn));
            }
            c[p] = cc;
        }
        __syncthreads();

        if (LAYER == 0) {
            // stream h_t to global fp16 buffer (vectorized from LDS)
            int row = tid >> 3, off = (tid & 7) * 32;
            const uint4* src = (const uint4*)(hwrite + row * 256 + off);
            uint4* dst = (uint4*)(h0out + ((size_t)(bbase + row) * 96 + t) * 256 + off);
#pragma unroll
            for (int jj = 0; jj < 4; ++jj) dst[jj] = src[jj];
        } else if (t >= 63) {
            // WMMA output projection: out = h_t @ WoutT + b_out  (waves 0..1, m = w)
            if (w < 2) {
                v8f acc;
                float bv = (lr < 8) ? bout[lr] : 0.0f;
#pragma unroll
                for (int e = 0; e < 8; ++e) acc[e] = bv;
                const _Float16* hrow = hwrite + (w * 16 + lr) * 256 + k0b;
                for (int kt = 0; kt < 8; ++kt) {
                    U16h ua;
                    ua.q[0] = *(const uint4*)(hrow + kt * 32);
                    ua.q[1] = *(const uint4*)(hrow + kt * 32 + 16);
                    const _Float16* bp = Wp + ((size_t)kt << 9) + l * 16;
                    U16h ub;
                    ub.q[0] = *(const uint4*)(bp);
                    ub.q[1] = *(const uint4*)(bp + 8);
                    acc = __builtin_amdgcn_wmma_f32_16x16x32_f16(
                        false, ua.v, false, ub.v, (short)0, acc, false, false);
                }
                if (lr < 8) {
                    const int rb = w * 16 + ((l >> 4) * 8);
#pragma unroll
                    for (int e = 0; e < 8; ++e)
                        out[((size_t)(bbase + rb + e) * 33 + (t - 63)) * 8 + lr] = acc[e];
                }
            }
        }
        // next iter: x-staging touches xbuf only; compute writes hbuf[pb] which no one reads
    }
}

// ---------------------------------------------------------------------------
// Host-side launcher
// ---------------------------------------------------------------------------
extern "C" void kernel_launch(void* const* d_in, const int* in_sizes, int n_in,
                              void* d_out, int out_size, void* d_ws, size_t ws_size,
                              hipStream_t stream) {
    const float* batch_init = (const float*)d_in[0];
    const float* batch_next = (const float*)d_in[1];
    const float* W_ih_l0 = (const float*)d_in[2];
    const float* W_hh_l0 = (const float*)d_in[3];
    const float* b_ih_l0 = (const float*)d_in[4];
    const float* b_hh_l0 = (const float*)d_in[5];
    const float* W_ih_l1 = (const float*)d_in[6];
    const float* W_hh_l1 = (const float*)d_in[7];
    const float* b_ih_l1 = (const float*)d_in[8];
    const float* b_hh_l1 = (const float*)d_in[9];
    const float* W_out   = (const float*)d_in[10];
    const float* b_out   = (const float*)d_in[11];
    float* out = (float*)d_out;

    // workspace layout
    char* ws = (char*)d_ws;
    const size_t WC0_BYTES = (size_t)288 * 1024 * 2;   // 589824
    const size_t WC1_BYTES = (size_t)512 * 1024 * 2;   // 1048576
    _Float16* Wc0 = (_Float16*)(ws);
    _Float16* Wc1 = (_Float16*)(ws + WC0_BYTES);
    _Float16* Wp  = (_Float16*)(ws + WC0_BYTES + WC1_BYTES);            // 8192 B
    float* bc0 = (float*)(ws + WC0_BYTES + WC1_BYTES + 8192);
    float* bc1 = (float*)(ws + WC0_BYTES + WC1_BYTES + 8192 + 4096);
    _Float16* h0buf = (_Float16*)(ws + WC0_BYTES + WC1_BYTES + 8192 + 8192); // (1024,96,256) fp16

    // weight prep
    {
        int total0 = 9 * 64 * 512;    // K=288 -> KT=9
        prep_wc_kernel<<<(total0 + 255) / 256, 256, 0, stream>>>(
            W_ih_l0, W_hh_l0, b_ih_l0, b_hh_l0, Wc0, bc0, 32, 9, 8, total0);
        int total1 = 16 * 64 * 512;   // K=512 -> KT=16
        prep_wc_kernel<<<(total1 + 255) / 256, 256, 0, stream>>>(
            W_ih_l1, W_hh_l1, b_ih_l1, b_hh_l1, Wc1, bc1, 256, 16, 256, total1);
        prep_wout_kernel<<<16, 256, 0, stream>>>(W_out, Wp);
    }

    // layer 0: inputs from batch_init/batch_next, h streamed to h0buf
    lstm_layer_kernel<32, 9, 0><<<32, 256, 0, stream>>>(
        batch_init, batch_next, nullptr, Wc0, bc0, h0buf, nullptr, nullptr, nullptr);

    // layer 1: inputs from h0buf, fused WMMA output projection into d_out
    lstm_layer_kernel<256, 16, 1><<<32, 256, 0, stream>>>(
        nullptr, nullptr, h0buf, Wc1, bc1, nullptr, Wp, b_out, out);
}
